// CapsNet_72988674228346
// MI455X (gfx1250) — matrix operand order repaired and checked by reading, not compiled, not run on hardware
//
#include <hip/hip_runtime.h>
#include <hip/hip_bf16.h>
#include <math.h>

typedef __attribute__((ext_vector_type(16))) _Float16 v16h;
typedef __attribute__((ext_vector_type(8)))  _Float16 v8h;
typedef __attribute__((ext_vector_type(8)))  float    v8f;

#define BATCH 256
#define G_    32
#define PD_   8
#define OC_   10
#define OD_   16
#define GS_   36
#define K2    20736   // 81*256 (conv2 im2col K)

// ---- CDNA5 data-mover support (guarded; falls back gracefully) ------------
#if defined(__has_builtin)
#if __has_builtin(__builtin_amdgcn_tensor_load_to_lds) && __has_builtin(__builtin_amdgcn_s_wait_tensorcnt)
#define HAVE_TDM 1
#endif
#if __has_builtin(__builtin_amdgcn_global_load_async_to_lds_b128)
#define HAVE_ASYNC_LDS 1
#endif
#endif
#ifndef HAVE_TDM
#define HAVE_TDM 0
#endif
#ifndef HAVE_ASYNC_LDS
#define HAVE_ASYNC_LDS 0
#endif

#if HAVE_TDM
typedef __attribute__((ext_vector_type(4))) unsigned int u32x4;
typedef __attribute__((ext_vector_type(8))) int          i32x8;
typedef __attribute__((ext_vector_type(4))) int          i32x4;

// 2D TDM load: tile rows x (row_qw * 8B) from global (row stride row_stride_qw*8B)
// into LDS at byte offset lds_off. data_size = 8 bytes.  D# per ISA 08 §8.
__device__ __forceinline__ void tdm_load_2d(unsigned long long gaddr, unsigned lds_off,
                                            int rows, int row_qw, int row_stride_qw) {
    u32x4 g0;
    g0[0] = 1u;                                            // count=1, user mode
    g0[1] = lds_off;                                       // lds_addr
    g0[2] = (unsigned)(gaddr & 0xffffffffull);             // global_addr[95:64]
    g0[3] = (unsigned)((gaddr >> 32) & 0x1ffffffull)       // global_addr[120:96]
          | (2u << 30);                                    // type = 2 ("image")
    i32x8 g1;
    g1[0] = 3 << 16;                                       // wg_mask=0, data_size=3 (8B)
    g1[1] = (row_qw & 0xffff) << 16;                       // tensor_dim0 lo16
    g1[2] = (rows & 0xffff) << 16;                         // tensor_dim0 hi=0 | tensor_dim1 lo16
    g1[3] = (row_qw & 0xffff) << 16;                       // tensor_dim1 hi=0 | tile_dim0
    g1[4] = (rows & 0xffff);                               // tile_dim1 | tile_dim2=0
    g1[5] = row_stride_qw;                                 // tensor_dim0_stride lo32
    g1[6] = 0;                                             // stride hi | dim1_stride lo
    g1[7] = 0;                                             // dim1_stride hi
    i32x4 z4 = {0, 0, 0, 0};                               // groups 2/3 unused (2D)
    i32x8 z8 = {0, 0, 0, 0, 0, 0, 0, 0};                   // extra group (6-arg form)
    __builtin_amdgcn_tensor_load_to_lds(g0, g1, z4, z4, z8, 0);
}
#endif

#if HAVE_ASYNC_LDS
typedef __attribute__((ext_vector_type(4))) int v4i_t;           // 16-byte payload
typedef __attribute__((address_space(1))) v4i_t* gp128_t;        // global src
typedef __attribute__((address_space(3))) v4i_t* lp128_t;        // LDS dst
__device__ __forceinline__ void async_copy16(const void* gsrc, void* ldst) {
    __builtin_amdgcn_global_load_async_to_lds_b128(
        (gp128_t)gsrc, (lp128_t)ldst, 0, 0);
}
__device__ __forceinline__ void wait_async0() {
#if __has_builtin(__builtin_amdgcn_s_wait_asynccnt)
    __builtin_amdgcn_s_wait_asynccnt(0);
#else
    asm volatile("s_wait_asynccnt 0x0" ::: "memory");
#endif
}
#endif

// ---------------------------------------------------------------------------
// Weight repack: W1 (256,1,9,9) f32 -> W1h [n=256][k=96] f16 (k>=81 zero)
// ---------------------------------------------------------------------------
__global__ void prep_w1(const float* __restrict__ W1, _Float16* __restrict__ W1h) {
    int idx = blockIdx.x * blockDim.x + threadIdx.x;   // 256*96
    if (idx >= 256 * 96) return;
    int n = idx / 96, k = idx % 96;
    W1h[idx] = (k < 81) ? (_Float16)W1[n * 81 + k] : (_Float16)0.0f;
}

// W2 (256,256,9,9) f32 -> W2h [o=256][k=20736] f16 with k = (ky*9+kx)*256 + c
__global__ void prep_w2(const float* __restrict__ W2, _Float16* __restrict__ W2h) {
    int idx = blockIdx.x * blockDim.x + threadIdx.x;   // 256*20736
    if (idx >= 256 * K2) return;
    int o = idx / K2, k = idx % K2;
    int pp = k >> 8, c = k & 255;                      // pp = ky*9+kx
    W2h[idx] = (_Float16)W2[(o * 256 + c) * 81 + pp];
}

// ---------------------------------------------------------------------------
// conv1 + bias + ReLU via WMMA.  GEMM: M=102400 (b,oy,ox), N=256, K=96.
// Output xh: f16 NHWC (B,20,20,256) == xh[m*256 + n].
// Per wave: 16M x 64N tile. Block = 4 waves -> 4 M-tiles. grid (1600, 4).
// ---------------------------------------------------------------------------
__global__ __launch_bounds__(128) void conv1_wmma(
    const float* __restrict__ input, const _Float16* __restrict__ W1h,
    const float* __restrict__ b1, _Float16* __restrict__ xh)
{
    const int lane = threadIdx.x & 31;
    const int wave = threadIdx.x >> 5;
    const int mt   = blockIdx.x * 4 + wave;      // 0..6399
    const int nb   = blockIdx.y * 64;            // 0/64/128/192
    const int mbase = mt * 16;

    // A-fragment row for this lane (16-bit A 16x32 layout)
    const int rowm = mbase + (lane & 15);
    const int bimg = rowm / 400;
    const int p    = rowm % 400;
    const int oy   = p / 20, ox = p % 20;
    const float* inp = input + bimg * 784 + oy * 28 + ox;
    const int klo  = (lane < 16) ? 0 : 8;
    const int koff = (lane < 16) ? 0 : 16;
    const int col  = lane & 15;

    v8f acc[4] = {};
    for (int kb = 0; kb < 96; kb += 32) {
        v16h a;
#pragma unroll
        for (int j = 0; j < 8; ++j) {
            int k0 = kb + klo + j;
            int k1 = kb + 16 + klo + j;
            float f0 = (k0 < 81) ? inp[(k0 / 9) * 28 + (k0 % 9)] : 0.0f;
            float f1 = (k1 < 81) ? inp[(k1 / 9) * 28 + (k1 % 9)] : 0.0f;
            a[j]     = (_Float16)f0;
            a[8 + j] = (_Float16)f1;
        }
#pragma unroll
        for (int i = 0; i < 4; ++i) {
            const _Float16* bp = W1h + (nb + i * 16 + col) * 96 + kb + klo;
            v8h blo = *(const v8h*)bp;
            v8h bhi = *(const v8h*)(bp + 16);
            v16h bf = __builtin_shufflevector(blo, bhi,
                0,1,2,3,4,5,6,7,8,9,10,11,12,13,14,15);
            acc[i] = __builtin_amdgcn_wmma_f32_16x16x32_f16(
                false, a, false, bf, (short)0, acc[i], false, false);
        }
    }
    // Epilogue: bias + relu + f16 store (C/D layout: lane<16 rows 0..7, else 8..15)
    const int rbase = (lane < 16) ? 0 : 8;
#pragma unroll
    for (int i = 0; i < 4; ++i) {
        int n = nb + i * 16 + col;
        float bias = b1[n];
#pragma unroll
        for (int r = 0; r < 8; ++r) {
            int m = mbase + rbase + r;
            float v = acc[i][r] + bias;
            v = v > 0.0f ? v : 0.0f;
            xh[(size_t)m * 256 + n] = (_Float16)v;
        }
    }
}

// ---------------------------------------------------------------------------
// conv2 + bias via WMMA, staging the shared B (weight) patch in LDS.
// GEMM: M=9216 (b,s), N=256, K=20736.  y: f32 (B,36,256) == y[m*256 + n].
// Per (ky,kx) patch: 64 cols x 256 K f16 = 32 KB, fetched once per block via
// TDM tensor_load_to_lds (TENSORcnt) -- or async-to-LDS / sync fallback.
// Per wave: 16M x 64N tile; block = 8 waves. grid (72, 4).
// Bs is DYNAMIC shared memory so its LDS byte offset is 0 for the TDM D#.
// ---------------------------------------------------------------------------
__global__ __launch_bounds__(256) void conv2_wmma(
    const _Float16* __restrict__ xh, const _Float16* __restrict__ W2h,
    const float* __restrict__ b2, float* __restrict__ y)
{
    extern __shared__ _Float16 Bs[];             // 32 KB, LDS offset 0

    const int tid  = threadIdx.x;
    const int lane = tid & 31;
    const int wave = tid >> 5;
    const int mt   = blockIdx.x * 8 + wave;      // 0..575
    const int nb   = blockIdx.y * 64;
    const int mbase = mt * 16;

    const int rowm = mbase + (lane & 15);
    const int bimg = rowm / 36;
    const int s    = rowm % 36;
    const int oy = s / 6, ox = s % 6;
    const _Float16* xbase = xh + (size_t)((bimg * 20 + oy * 2) * 20 + ox * 2) * 256;
    const int klo  = (lane < 16) ? 0 : 8;
    const int koff = (lane < 16) ? 0 : 16;
    const int col  = lane & 15;

    v8f acc[4] = {};
    for (int pp = 0; pp < 81; ++pp) {
        const int ky = pp / 9, kx = pp % 9;

        __syncthreads();                          // previous patch fully consumed
#if HAVE_TDM
        if (wave == 0) {
            unsigned long long gaddr = (unsigned long long)W2h
                + (unsigned long long)nb * (K2 * 2) + (unsigned long long)pp * 512;
            tdm_load_2d(gaddr, 0u, /*rows=*/64, /*row_qw=*/64,
                        /*row_stride_qw=*/K2 * 2 / 8);    // 5184 qwords = 41472 B
            __builtin_amdgcn_s_wait_tensorcnt(0);
        }
#else
        // Stage B patch: W2h[(nb+c)*K2 + pp*256 + k] -> Bs[c*256 + k]
#pragma unroll
        for (int i = 0; i < 8; ++i) {
            int off = tid * 128 + i * 16;         // byte offset within tile
            int c   = off >> 9;                   // / 512 (row = one col's 256 K)
            int kb  = off & 511;                  // byte within row
            const char* src = (const char*)W2h
                + ((size_t)(nb + c) * K2) * 2 + (size_t)pp * 512 + kb;
            char* dst = (char*)Bs + off;
#if HAVE_ASYNC_LDS
            async_copy16(src, dst);
#else
            *(float4*)dst = *(const float4*)src;
#endif
        }
#if HAVE_ASYNC_LDS
        wait_async0();
#endif
#endif
        __syncthreads();                          // staged patch visible to all waves

        const _Float16* arow = xbase + (ky * 20 + kx) * 256;
        for (int c0 = 0; c0 < 256; c0 += 32) {
            const _Float16* ap = arow + c0 + klo;
            v8h alo = *(const v8h*)ap;
            v8h ahi = *(const v8h*)(ap + 16);
            v16h a = __builtin_shufflevector(alo, ahi,
                0,1,2,3,4,5,6,7,8,9,10,11,12,13,14,15);
#pragma unroll
            for (int i = 0; i < 4; ++i) {
                const _Float16* bp = Bs + (i * 16 + col) * 256 + c0 + klo;
                v8h blo = *(const v8h*)bp;
                v8h bhi = *(const v8h*)(bp + 16);
                v16h bf = __builtin_shufflevector(blo, bhi,
                    0,1,2,3,4,5,6,7,8,9,10,11,12,13,14,15);
                acc[i] = __builtin_amdgcn_wmma_f32_16x16x32_f16(
                    false, a, false, bf, (short)0, acc[i], false, false);
            }
        }
    }
    const int rbase = (lane < 16) ? 0 : 8;
#pragma unroll
    for (int i = 0; i < 4; ++i) {
        int n = nb + i * 16 + col;
        float bias = b2[n];
#pragma unroll
        for (int r = 0; r < 8; ++r) {
            int m = mbase + rbase + r;
            y[(size_t)m * 256 + n] = acc[i][r] + bias;
        }
    }
}

// ---------------------------------------------------------------------------
// Fused squash + capsule predictions + 3 routing iterations, per (b,g).
// grid (256, 32), 160 threads. Everything lives in ~28 KB LDS.
// ---------------------------------------------------------------------------
__global__ __launch_bounds__(160) void routing_kernel(
    const float* __restrict__ y, const float* __restrict__ Wcaps,
    const float* __restrict__ b_route, float* __restrict__ vpart)
{
    __shared__ float uS[GS_ * PD_];          // 288
    __shared__ float upS[GS_ * OC_ * OD_];   // 5760
    __shared__ float blogS[GS_ * OC_];       // 360
    __shared__ float cS[GS_ * OC_];          // 360
    __shared__ float sS[OC_ * OD_];          // 160
    __shared__ float vS[OC_ * OD_];          // 160
    __shared__ float fS[GS_];                // squash factors (36 / 10 reused)

    const int b = blockIdx.x;
    const int g = blockIdx.y;
    const int tid = threadIdx.x;

    // u_raw[s][d] = y[b, s, g*8+d]
    for (int idx = tid; idx < GS_ * PD_; idx += 160) {
        int s = idx / PD_, d = idx % PD_;
        uS[idx] = y[((size_t)b * GS_ + s) * 256 + g * PD_ + d];
    }
    __syncthreads();
    if (tid < GS_) {
        float l2 = 0.0f;
        for (int d = 0; d < PD_; ++d) { float t = uS[tid * PD_ + d]; l2 += t * t; }
        float l = sqrtf(l2);
        fS[tid] = (l2 / (1.0f + l2)) / (l + 1e-8f);
    }
    __syncthreads();
    for (int idx = tid; idx < GS_ * PD_; idx += 160) uS[idx] *= fS[idx / PD_];
    __syncthreads();

    // up[s][j] = sum_i u[s][i] * Wcaps[g][s][i][j]
    const float* Wc = Wcaps + (size_t)g * GS_ * PD_ * (OC_ * OD_);
    for (int idx = tid; idx < GS_ * OC_ * OD_; idx += 160) {
        int s = idx / (OC_ * OD_), j = idx % (OC_ * OD_);
        float a = 0.0f;
#pragma unroll
        for (int i = 0; i < PD_; ++i)
            a += uS[s * PD_ + i] * Wc[(s * PD_ + i) * (OC_ * OD_) + j];
        upS[idx] = a;
    }
    for (int idx = tid; idx < GS_ * OC_; idx += 160)
        blogS[idx] = b_route[g * GS_ * OC_ + idx];
    __syncthreads();

    for (int r = 0; r < 3; ++r) {
        if (tid < GS_) {
            float mx = -1e30f;
            for (int o = 0; o < OC_; ++o) mx = fmaxf(mx, blogS[tid * OC_ + o]);
            float e[OC_]; float sum = 0.0f;
            for (int o = 0; o < OC_; ++o) { e[o] = expf(blogS[tid * OC_ + o] - mx); sum += e[o]; }
            float inv = 1.0f / sum;
            for (int o = 0; o < OC_; ++o) cS[tid * OC_ + o] = e[o] * inv;
        }
        __syncthreads();
        {   // s[o][d] = sum_s c[s][o] * up[s][o*16+d]   (tid = o*16+d)
            int o = tid / OD_;
            float a = 0.0f;
            for (int s = 0; s < GS_; ++s)
                a += cS[s * OC_ + o] * upS[s * (OC_ * OD_) + tid];
            sS[tid] = a;
        }
        __syncthreads();
        if (tid < OC_) {
            float l2 = 0.0f;
            for (int d = 0; d < OD_; ++d) { float t = sS[tid * OD_ + d]; l2 += t * t; }
            float l = sqrtf(l2);
            fS[tid] = (l2 / (1.0f + l2)) / (l + 1e-8f);
        }
        __syncthreads();
        vS[tid] = sS[tid] * fS[tid / OD_];
        __syncthreads();
        if (r < 2) {
            for (int idx = tid; idx < GS_ * OC_; idx += 160) {
                int s = idx / OC_, o = idx % OC_;
                float dot = 0.0f;
#pragma unroll
                for (int d = 0; d < OD_; ++d)
                    dot += upS[s * (OC_ * OD_) + o * OD_ + d] * vS[o * OD_ + d];
                blogS[idx] += dot;
            }
            __syncthreads();
        }
    }
    vpart[((size_t)b * G_ + g) * (OC_ * OD_) + tid] = vS[tid];
}

// ---------------------------------------------------------------------------
// Final: v[b] = sum_g vpart[b][g]; probs[b][o] = ||v[b,o,:]||
// ---------------------------------------------------------------------------
__global__ __launch_bounds__(160) void finalize_kernel(
    const float* __restrict__ vpart, float* __restrict__ out)
{
    __shared__ float vv[OC_ * OD_];
    const int b = blockIdx.x, tid = threadIdx.x;
    float acc = 0.0f;
    for (int g = 0; g < G_; ++g)
        acc += vpart[((size_t)b * G_ + g) * (OC_ * OD_) + tid];
    out[(size_t)b * (OC_ * OD_) + tid] = acc;
    vv[tid] = acc;
    __syncthreads();
    if (tid < OC_) {
        float l2 = 0.0f;
        for (int d = 0; d < OD_; ++d) { float t = vv[tid * OD_ + d]; l2 += t * t; }
        out[(size_t)BATCH * (OC_ * OD_) + b * OC_ + tid] = sqrtf(l2);
    }
}

// ---------------------------------------------------------------------------
extern "C" void kernel_launch(void* const* d_in, const int* in_sizes, int n_in,
                              void* d_out, int out_size, void* d_ws, size_t ws_size,
                              hipStream_t stream) {
    const float* input   = (const float*)d_in[0];
    const float* W1      = (const float*)d_in[1];
    const float* b1      = (const float*)d_in[2];
    const float* W2      = (const float*)d_in[3];
    const float* b2      = (const float*)d_in[4];
    const float* Wcaps   = (const float*)d_in[5];
    const float* b_route = (const float*)d_in[6];
    float* out = (float*)d_out;

    char* ws = (char*)d_ws;
    size_t off = 0;
    auto take = [&](size_t bytes) -> char* {
        char* p = ws + off;
        off = (off + bytes + 255) & ~(size_t)255;
        return p;
    };
    _Float16* W1h  = (_Float16*)take((size_t)256 * 96 * 2);       //  48 KB
    _Float16* W2h  = (_Float16*)take((size_t)256 * K2 * 2);       //  10.6 MB
    _Float16* xh   = (_Float16*)take((size_t)102400 * 256 * 2);   //  52.4 MB
    float*    yint = (float*)   take((size_t)9216 * 256 * 4);     //   9.4 MB
    float*    vpart= (float*)   take((size_t)256 * G_ * 160 * 4); //   5.2 MB

    prep_w1<<<(256 * 96 + 255) / 256, 256, 0, stream>>>(W1, W1h);
    prep_w2<<<(256 * K2 + 255) / 256, 256, 0, stream>>>(W2, W2h);
    conv1_wmma<<<dim3(1600, 4), 128, 0, stream>>>(input, W1h, b1, xh);
    conv2_wmma<<<dim3(72, 4), 256, 32768, stream>>>(xh, W2h, b2, yint);
    routing_kernel<<<dim3(BATCH, G_), 160, 0, stream>>>(yint, Wcaps, b_route, vpart);
    finalize_kernel<<<BATCH, 160, 0, stream>>>(vpart, out);
}
